// BioAegisXAlpha_77730318123350
// MI455X (gfx1250) — matrix operand order repaired and checked by reading, not compiled, not run on hardware
//
#include <hip/hip_runtime.h>
#include <math.h>
#include <stdint.h>

// ---------------- problem sizes ----------------
#define NV   131072   // nodes
#define NE   524288   // edges
#define NG   4096     // graphs
#define HD   128      // hidden
#define IND  11
#define EDD  5
#define FPDIM 2048
#define DDD  12
#define NOUTC 13
#define NEGS 0.01f
#define BNEPS 1e-5f

typedef __attribute__((ext_vector_type(16))) __bf16 v16bf;
typedef __attribute__((ext_vector_type(8)))  float  v8f;

// float -> bf16 round-to-nearest-even (integer ops)
__device__ __forceinline__ __bf16 f2bf(float f) {
  unsigned u = __float_as_uint(f);
  u = u + 0x7fffu + ((u >> 16) & 1u);
  unsigned short h = (unsigned short)(u >> 16);
  __bf16 b;
  __builtin_memcpy(&b, &h, sizeof(h));
  return b;
}

// CDNA5 async global->LDS copy (16B per lane), tracked by ASYNCcnt.
// Low 32 bits of a generic pointer to __shared__ data are the LDS offset.
__device__ __forceinline__ void async_copy_b128(void* lds, const void* gsrc) {
  uint32_t la = (uint32_t)(uintptr_t)lds;
  uint64_t ga = (uint64_t)(uintptr_t)gsrc;
  asm volatile("global_load_async_to_lds_b128 %0, %1, off"
               :: "v"(la), "v"(ga) : "memory");
}
__device__ __forceinline__ void wait_async0() {
  asm volatile("s_wait_asynccnt 0" ::: "memory");
}

// activation: 0=none 1=relu 2=leaky(0.01) 3=elu
template<int ACT> __device__ __forceinline__ float actf(float x) {
  if constexpr (ACT == 1) return x > 0.f ? x : 0.f;
  else if constexpr (ACT == 2) return x > 0.f ? x : NEGS * x;
  else if constexpr (ACT == 3) return x > 0.f ? x : (__expf(x) - 1.f);
  else return x;
}

// ---- WMMA fragment loaders (CDNA5 16-bit layouts, ISA 7.12.2) ----
__device__ __forceinline__ v16bf load_a_frag(const __bf16* A, int stride, int row0, int k0, int lane) {
  int lr = lane & 15, half = lane >> 4;
  const __bf16* p = A + (row0 + lr) * stride + k0 + half * 8;
  v16bf a;
#pragma unroll
  for (int i = 0; i < 8; ++i) a[i] = p[i];
#pragma unroll
  for (int i = 0; i < 8; ++i) a[8 + i] = p[16 + i];
  return a;
}
__device__ __forceinline__ v16bf load_b_frag(const __bf16* Bt, int stride, int col0, int k0, int lane) {
  int lr = lane & 15, half = lane >> 4;
  const __bf16* p = Bt + (col0 + lr) * stride + k0 + half * 8;
  v16bf b;
#pragma unroll
  for (int i = 0; i < 8; ++i) b[i] = p[i];
#pragma unroll
  for (int i = 0; i < 8; ++i) b[8 + i] = p[16 + i];
  return b;
}

// ---------------- generic GEMM: OUT[m*ldo+n] = act(A[M,K](f32) @ Wb[N,K](bf16)^T + bias) ----
// grid = (Nout/64, M/64), block = 256 (8 waves). K multiple of 32.
// B tiles staged via async global->LDS (raw bf16); A tiles converted f32->bf16 in VALU.
template<int ACT>
__global__ __launch_bounds__(256)
void gemm_bias_act(const float* __restrict__ A, const __bf16* __restrict__ Wb,
                   const float* __restrict__ bias, float* __restrict__ OUT,
                   int ldo, int K) {
  __shared__ __align__(16) __bf16 As[64 * 40];
  __shared__ __align__(16) __bf16 Bs[64 * 40];
  const int row0 = blockIdx.y * 64;
  const int col0 = blockIdx.x * 64;
  const int tid = threadIdx.x;
  const int lane = tid & 31;
  const int wave = tid >> 5;
  const int wy = wave & 3;   // row tile (16 rows)
  const int wx = wave >> 2;  // col half (32 cols)
  v8f acc0 = {}; v8f acc1 = {};
  const int bn = tid >> 2, bc = tid & 3;   // B staging: one b128 per thread
  for (int kt = 0; kt < K; kt += 32) {
    async_copy_b128(&Bs[bn * 40 + bc * 8], Wb + (size_t)(col0 + bn) * K + kt + bc * 8);
    for (int i = tid; i < 64 * 32; i += 256) {
      int r = i >> 5, c = i & 31;
      As[r * 40 + c] = f2bf(A[(size_t)(row0 + r) * K + kt + c]);
    }
    wait_async0();
    __syncthreads();
    v16bf a  = load_a_frag(As, 40, wy * 16, 0, lane);
    v16bf b0 = load_b_frag(Bs, 40, wx * 32, 0, lane);
    v16bf b1 = load_b_frag(Bs, 40, wx * 32 + 16, 0, lane);
    acc0 = __builtin_amdgcn_wmma_f32_16x16x32_bf16(false, a, false, b0, (short)0, acc0, false, false);
    acc1 = __builtin_amdgcn_wmma_f32_16x16x32_bf16(false, a, false, b1, (short)0, acc1, false, false);
    __syncthreads();
  }
  const int lr = lane & 15, half = lane >> 4;
#pragma unroll
  for (int r = 0; r < 8; ++r) {
    int m = row0 + wy * 16 + r + half * 8;
    int n0 = col0 + wx * 32 + lr;
    int n1 = n0 + 16;
    float v0 = acc0[r] + (bias ? bias[n0] : 0.f);
    float v1 = acc1[r] + (bias ? bias[n1] : 0.f);
    OUT[(size_t)m * ldo + n0] = actf<ACT>(v0);
    OUT[(size_t)m * ldo + n1] = actf<ACT>(v1);
  }
}

// ---------------- GATEConv fused edge-logit kernel ----------------
// ej = leaky([x[src]||edge_attr] @ W1^T); logit = leaky(ej . att_l + sdst[dst])
// grid = NE/64, block = 256. K = 133 padded to 160. x rows gathered async from bf16 mirror.
__global__ __launch_bounds__(256)
void gate_edge_logits(const __bf16* __restrict__ hxb, const float* __restrict__ eattr,
                      const int* __restrict__ src, const int* __restrict__ dst,
                      const __bf16* __restrict__ W1b, const float* __restrict__ att_l,
                      const float* __restrict__ sdst, float* __restrict__ logits) {
  __shared__ __align__(16) __bf16 As[64 * 160];
  __shared__ __align__(16) __bf16 Bs[128 * 160];
  __shared__ float sAcc[64];
  const int e0 = blockIdx.x * 64;
  const int tid = threadIdx.x;
  const int lane = tid & 31;
  const int wave = tid >> 5;
  if (tid < 64) sAcc[tid] = 0.f;
  // gather x[src] rows (128 cols = 16 b128 chunks per row) asynchronously
  for (int ch = tid; ch < 64 * 16; ch += 256) {
    int r = ch >> 4, c = ch & 15;
    async_copy_b128(&As[r * 160 + c * 8], hxb + (size_t)src[e0 + r] * HD + c * 8);
  }
  // W1 (bf16) with zero padding 133 -> 160
  for (int i = tid; i < 128 * 160; i += 256) {
    int n = i / 160, k = i - n * 160;
    Bs[i] = (k < 133) ? W1b[n * 133 + k] : f2bf(0.f);
  }
  // edge_attr + pad columns 128..159
  for (int i = tid; i < 64 * 32; i += 256) {
    int r = i >> 5, k = (i & 31) + 128;
    int e = e0 + r;
    As[r * 160 + k] = (k < 133) ? f2bf(eattr[(size_t)e * EDD + (k - 128)]) : f2bf(0.f);
  }
  wait_async0();
  __syncthreads();
  const int cb = wave * 16;
  const int lr = lane & 15, half = lane >> 4;
  const float al = att_l[cb + lr];
  for (int rt = 0; rt < 4; ++rt) {
    v8f acc = {};
#pragma unroll
    for (int kc = 0; kc < 5; ++kc) {
      v16bf a = load_a_frag(As, 160, rt * 16, kc * 32, lane);
      v16bf b = load_b_frag(Bs, 160, cb, kc * 32, lane);
      acc = __builtin_amdgcn_wmma_f32_16x16x32_bf16(false, a, false, b, (short)0, acc, false, false);
    }
#pragma unroll
    for (int r = 0; r < 8; ++r) {
      float ej = actf<2>(acc[r]);
      atomicAdd(&sAcc[rt * 16 + r + half * 8], ej * al);
    }
  }
  __syncthreads();
  if (tid < 64) {
    int e = e0 + tid;
    logits[e] = actf<2>(sAcc[tid] + sdst[dst[e]]);
  }
}

// ---------------- GATEConv fused message kernel ----------------
// out[dst] += (x[src] @ W2^T) * alpha    (alpha = ev/(ssum[dst]+1e-16))
__global__ __launch_bounds__(256)
void gate_msg_scatter(const __bf16* __restrict__ hxb, const int* __restrict__ src,
                      const int* __restrict__ dst, const __bf16* __restrict__ W2b,
                      const float* __restrict__ ev, const float* __restrict__ ssum,
                      float* __restrict__ out) {
  __shared__ __align__(16) __bf16 As[64 * 128];
  __shared__ __align__(16) __bf16 Bs[128 * 128];
  const int e0 = blockIdx.x * 64;
  const int tid = threadIdx.x;
  const int lane = tid & 31;
  const int wave = tid >> 5;
  for (int ch = tid; ch < 2048; ch += 256)          // W2: 32KB straight async copy
    async_copy_b128(&Bs[ch * 8], W2b + ch * 8);
  for (int ch = tid; ch < 64 * 16; ch += 256) {     // gathered x[src] rows
    int r = ch >> 4, c = ch & 15;
    async_copy_b128(&As[r * 128 + c * 8], hxb + (size_t)src[e0 + r] * HD + c * 8);
  }
  wait_async0();
  __syncthreads();
  const int cb = wave * 16;
  const int lr = lane & 15, half = lane >> 4;
  for (int rt = 0; rt < 4; ++rt) {
    v8f acc = {};
#pragma unroll
    for (int kc = 0; kc < 4; ++kc) {
      v16bf a = load_a_frag(As, 128, rt * 16, kc * 32, lane);
      v16bf b = load_b_frag(Bs, 128, cb, kc * 32, lane);
      acc = __builtin_amdgcn_wmma_f32_16x16x32_bf16(false, a, false, b, (short)0, acc, false, false);
    }
#pragma unroll
    for (int r = 0; r < 8; ++r) {
      int e = e0 + rt * 16 + r + half * 8;
      float alpha = ev[e] / (ssum[dst[e]] + 1e-16f);
      atomicAdd(&out[(size_t)dst[e] * HD + cb + lr], acc[r] * alpha);
    }
  }
}

// ---------------- small kernels ----------------
__global__ void fill_f(float* __restrict__ p, float v, int n) {
  int i = blockIdx.x * 256 + threadIdx.x;
  if (i < n) p[i] = v;
}

__global__ void cvt_bf16(const float* __restrict__ s, __bf16* __restrict__ d, int n) {
  int i = blockIdx.x * 256 + threadIdx.x;
  if (i < n) d[i] = f2bf(s[i]);
}

__global__ void lin1_kernel(const float* __restrict__ x, const float* __restrict__ W,
                            const float* __restrict__ b, float* __restrict__ hx,
                            __bf16* __restrict__ hxb) {
  __shared__ float xr[IND];
  int node = blockIdx.x, j = threadIdx.x;
  if (j < IND) xr[j] = x[node * IND + j];
  __syncthreads();
  float s = b[j];
#pragma unroll
  for (int k = 0; k < IND; ++k) s += xr[k] * W[j * IND + k];
  float v = actf<2>(s);
  hx[(size_t)node * HD + j] = v;
  hxb[(size_t)node * HD + j] = f2bf(v);
}

// per-row dot with one or two 128-vectors; grid=M, block=128
__global__ void row_dots(const float* __restrict__ X, const float* __restrict__ v1,
                         const float* __restrict__ v2, float* __restrict__ o1,
                         float* __restrict__ o2) {
  __shared__ float s1[128];
  __shared__ float s2[128];
  int row = blockIdx.x, t = threadIdx.x;
  float xv = X[(size_t)row * HD + t];
  s1[t] = xv * v1[t];
  s2[t] = v2 ? xv * v2[t] : 0.f;
  __syncthreads();
  for (int off = 64; off > 0; off >>= 1) {
    if (t < off) { s1[t] += s1[t + off]; s2[t] += s2[t + off]; }
    __syncthreads();
  }
  if (t == 0) { o1[row] = s1[0]; if (o2) o2[row] = s2[0]; }
}

__device__ __forceinline__ void atomicMaxF(float* addr, float val) {
  unsigned* p = (unsigned*)addr;
  unsigned old = *p;
  while (__uint_as_float(old) < val) {
    unsigned assumed = old;
    old = atomicCAS(p, assumed, __float_as_uint(val));
    if (old == assumed) break;
  }
}

__global__ void seg_atomic_max(const float* __restrict__ lg, const int* __restrict__ seg,
                               float* __restrict__ m, int n) {
  int i = blockIdx.x * 256 + threadIdx.x;
  if (i < n) atomicMaxF(&m[seg[i]], lg[i]);
}

__global__ void seg_exp_sum(const float* __restrict__ lg, const int* __restrict__ seg,
                            const float* __restrict__ m, float* __restrict__ ev,
                            float* __restrict__ s, int n) {
  int i = blockIdx.x * 256 + threadIdx.x;
  if (i < n) {
    float mm = m[seg[i]];
    if (!__builtin_isfinite(mm)) mm = 0.f;
    float e = __expf(lg[i] - mm);
    ev[i] = e;
    atomicAdd(&s[seg[i]], e);
  }
}

__global__ void edge_logits(const float* __restrict__ a_src, const float* __restrict__ a_dst,
                            const int* __restrict__ src, const int* __restrict__ dst,
                            float* __restrict__ lg, int n) {
  int i = blockIdx.x * 256 + threadIdx.x;
  if (i < n) {
    int sidx = src ? src[i] : i;
    lg[i] = actf<2>(a_src[sidx] + a_dst[dst[i]]);
  }
}

__global__ void gat_scatter(const float* __restrict__ hs, const int* __restrict__ src,
                            const int* __restrict__ dst, const float* __restrict__ ev,
                            const float* __restrict__ ssum, float* __restrict__ out,
                            int n) {
  size_t idx = (size_t)blockIdx.x * 256 + threadIdx.x;
  if (idx >= (size_t)n * HD) return;
  int e = (int)(idx >> 7), c = (int)(idx & 127);
  int sidx = src ? src[e] : e;
  float alpha = ev[e] / (ssum[dst[e]] + 1e-16f);
  atomicAdd(&out[(size_t)dst[e] * HD + c], hs[(size_t)sidx * HD + c] * alpha);
}

__global__ void elu_bias(const float* __restrict__ xin, const float* __restrict__ b,
                         float* __restrict__ out, size_t n) {
  size_t i = (size_t)blockIdx.x * 256 + threadIdx.x;
  if (i < n) { int c = (int)(i & 127); out[i] = actf<3>(xin[i] + b[c]); }
}

// torch GRUCell math + relu
__global__ void gru_eltwise(const float* __restrict__ gi, const float* __restrict__ gh,
                            const float* __restrict__ hprev, float* __restrict__ out,
                            size_t rows) {
  size_t i = (size_t)blockIdx.x * 256 + threadIdx.x;
  if (i >= rows * HD) return;
  size_t row = i >> 7; int c = (int)(i & 127);
  const float* gir = gi + row * 384;
  const float* ghr = gh + row * 384;
  float r  = 1.f / (1.f + __expf(-(gir[c] + ghr[c])));
  float z  = 1.f / (1.f + __expf(-(gir[128 + c] + ghr[128 + c])));
  float ng = tanhf(gir[256 + c] + r * ghr[256 + c]);
  float v = (1.f - z) * ng + z * hprev[i];
  out[i] = v > 0.f ? v : 0.f;
}

__global__ void pool_kernel(const float* __restrict__ hx, const int* __restrict__ batch,
                            float* __restrict__ pooled) {
  size_t i = (size_t)blockIdx.x * 256 + threadIdx.x;
  if (i >= (size_t)NV * HD) return;
  size_t node = i >> 7; int c = (int)(i & 127);
  atomicAdd(&pooled[(size_t)batch[node] * HD + c], hx[i]);
}

__global__ void relu_inplace(float* __restrict__ p, size_t n) {
  size_t i = (size_t)blockIdx.x * 256 + threadIdx.x;
  if (i < n) p[i] = p[i] > 0.f ? p[i] : 0.f;
}

__global__ void bn_inplace(float* __restrict__ xp, const float* __restrict__ g,
                           const float* __restrict__ be, const float* __restrict__ m,
                           const float* __restrict__ v, int mask, size_t n) {
  size_t i = (size_t)blockIdx.x * 256 + threadIdx.x;
  if (i < n) {
    int c = (int)(i & mask);
    xp[i] = (xp[i] - m[c]) * rsqrtf(v[c] + BNEPS) * g[c] + be[c];
  }
}

__global__ void desc1_kernel(const float* __restrict__ d, const float* __restrict__ W,
                             const float* __restrict__ b, float* __restrict__ out) {
  __shared__ float dr[DDD];
  int g = blockIdx.x, j = threadIdx.x;  // block = 64
  if (j < DDD) dr[j] = d[g * DDD + j];
  __syncthreads();
  float s = b[j];
#pragma unroll
  for (int k = 0; k < DDD; ++k) s += dr[k] * W[j * DDD + k];
  out[g * 64 + j] = s > 0.f ? s : 0.f;
}

// logits [NG,13] then percentage [NG], concat flat in d_out
__global__ void heads_kernel(const float* __restrict__ sh, const float* __restrict__ toxW,
                             const float* __restrict__ toxb, const float* __restrict__ regW,
                             const float* __restrict__ regb, float* __restrict__ outp) {
  __shared__ float sr[256];
  int g = blockIdx.x, t = threadIdx.x;
  sr[t] = sh[(size_t)g * 256 + t];
  __syncthreads();
  if (t < NOUTC) {
    float s = toxb[t];
    for (int k = 0; k < 256; ++k) s += sr[k] * toxW[t * 256 + k];
    outp[(size_t)g * NOUTC + t] = s;
  } else if (t == NOUTC) {
    float s = regb[0];
    for (int k = 0; k < 256; ++k) s += sr[k] * regW[k];
    outp[(size_t)NG * NOUTC + g] = s;
  }
}

// ---------------- launcher ----------------
extern "C" void kernel_launch(void* const* d_in, const int* in_sizes, int n_in,
                              void* d_out, int out_size, void* d_ws, size_t ws_size,
                              hipStream_t stream) {
  (void)in_sizes; (void)n_in; (void)out_size; (void)ws_size;
  const float* x     = (const float*)d_in[0];
  const int*   ei    = (const int*)d_in[1];
  const int*   src   = ei;
  const int*   dst   = ei + NE;
  const float* eattr = (const float*)d_in[2];
  const int*   batch = (const int*)d_in[3];
  const float* fpr   = (const float*)d_in[4];
  const float* descr = (const float*)d_in[5];
#define PP(i) ((const float*)d_in[i])
  const float* lin1_W = PP(6);  const float* lin1_b = PP(7);
  const float* gW1 = PP(8);     const float* gW2 = PP(9);
  const float* g_attl = PP(10); const float* g_attr_ = PP(11); const float* g_b = PP(12);
  const float* g0Wih = PP(13);  const float* g0Whh = PP(14);
  const float* g0bih = PP(15);  const float* g0bhh = PP(16);
  const float* acW[2]   = {PP(17), PP(21)};
  const float* acAs[2]  = {PP(18), PP(22)};
  const float* acAd[2]  = {PP(19), PP(23)};
  const float* acB[2]   = {PP(20), PP(24)};
  const float* agWih[2] = {PP(25), PP(29)};
  const float* agWhh[2] = {PP(26), PP(30)};
  const float* agBih[2] = {PP(27), PP(31)};
  const float* agBhh[2] = {PP(28), PP(32)};
  const float* mcW = PP(33); const float* mcAs = PP(34);
  const float* mcAd = PP(35); const float* mcB = PP(36);
  const float* mgWih = PP(37); const float* mgWhh = PP(38);
  const float* mgBih = PP(39); const float* mgBhh = PP(40);
  const float* l2W = PP(41); const float* l2b = PP(42);
  const float* fpW1 = PP(43); const float* fpb1 = PP(44);
  const float* fpg = PP(45); const float* fpbe = PP(46);
  const float* fpm = PP(47); const float* fpv = PP(48);
  const float* fpW2 = PP(49); const float* fpb2 = PP(50);
  const float* deW1 = PP(51); const float* deb1 = PP(52);
  const float* deg = PP(53); const float* debe = PP(54);
  const float* dem = PP(55); const float* dev_ = PP(56);
  const float* deW2 = PP(57); const float* deb2 = PP(58);
  const float* fusW = PP(59); const float* fusb = PP(60);
  const float* toxW = PP(61); const float* toxb = PP(62);
  const float* regW = PP(63); const float* regb = PP(64);
#undef PP

  float* ws = (float*)d_ws;
  size_t off = 0;
  auto wsalloc = [&](size_t n) { float* p = ws + off; off += n; return p; };
  float* hx   = wsalloc((size_t)NV * HD);
  float* hbuf = wsalloc((size_t)NV * HD);
  float* agg  = wsalloc((size_t)NV * HD);
  float* hs   = wsalloc((size_t)NV * HD);
  float* gi   = wsalloc((size_t)NV * 384);
  float* gh   = wsalloc((size_t)NV * 384);
  float* d1   = wsalloc(NV);
  float* d2   = wsalloc(NV);
  float* sdst = wsalloc(NV);
  float* segm = wsalloc(NV);
  float* segs = wsalloc(NV);
  float* elog = wsalloc(NE);
  float* eexp = wsalloc(NE);
  float* molo = wsalloc((size_t)NG * HD);
  float* hd   = wsalloc((size_t)NG * HD);
  float* molh = wsalloc((size_t)NG * HD);
  float* magg = wsalloc((size_t)NG * HD);
  float* f512 = wsalloc((size_t)NG * 512);
  float* d64  = wsalloc((size_t)NG * 64);
  float* comb = wsalloc((size_t)NG * 384);
  float* shrd = wsalloc((size_t)NG * 256);
  // bf16 region (carved from the same workspace; float-sized slots, 2 bf16 per float)
  __bf16* wb = (__bf16*)(ws + off);
  size_t boff = 0;
  auto balloc = [&](size_t n) { __bf16* p = wb + boff; boff += (n + 7) & ~(size_t)7; return p; };
  __bf16* hxb    = balloc((size_t)NV * HD);
  __bf16* gW1b   = balloc(128 * 133);
  __bf16* gW2b   = balloc(128 * 128);
  __bf16* g0Wihb = balloc(384 * 128);
  __bf16* g0Whhb = balloc(384 * 128);
  __bf16* acWb[2]   = {balloc(128 * 128), balloc(128 * 128)};
  __bf16* agWihb[2] = {balloc(384 * 128), balloc(384 * 128)};
  __bf16* agWhhb[2] = {balloc(384 * 128), balloc(384 * 128)};
  __bf16* mcWb   = balloc(128 * 128);
  __bf16* mgWihb = balloc(384 * 128);
  __bf16* mgWhhb = balloc(384 * 128);
  __bf16* l2Wb   = balloc(128 * 128);
  __bf16* fpW1b  = balloc(512 * 2048);
  __bf16* fpW2b  = balloc(128 * 512);
  __bf16* deW2b  = balloc(128 * 64);
  __bf16* fusWb  = balloc(256 * 384);

  auto cvt = [&](const float* s, __bf16* d, int n) {
    cvt_bf16<<<(n + 255) / 256, 256, 0, stream>>>(s, d, n);
  };
  // pre-convert all GEMM weights to bf16 (once; L2-resident afterwards)
  cvt(gW1, gW1b, 128 * 133);       cvt(gW2, gW2b, 128 * 128);
  cvt(g0Wih, g0Wihb, 384 * 128);   cvt(g0Whh, g0Whhb, 384 * 128);
  for (int l = 0; l < 2; ++l) {
    cvt(acW[l], acWb[l], 128 * 128);
    cvt(agWih[l], agWihb[l], 384 * 128);
    cvt(agWhh[l], agWhhb[l], 384 * 128);
  }
  cvt(mcW, mcWb, 128 * 128);       cvt(mgWih, mgWihb, 384 * 128);
  cvt(mgWhh, mgWhhb, 384 * 128);   cvt(l2W, l2Wb, 128 * 128);
  cvt(fpW1, fpW1b, 512 * 2048);    cvt(fpW2, fpW2b, 128 * 512);
  cvt(deW2, deW2b, 128 * 64);      cvt(fusW, fusWb, 256 * 384);

  auto fill = [&](float* p, float v, size_t n) {
    fill_f<<<(unsigned)((n + 255) / 256), 256, 0, stream>>>(p, v, (int)n);
  };
  auto softmax_seg = [&](const float* lg, const int* seg, int nEl, int nSeg, float* ev) {
    fill(segm, -INFINITY, nSeg);
    fill(segs, 0.f, nSeg);
    seg_atomic_max<<<(nEl + 255) / 256, 256, 0, stream>>>(lg, seg, segm, nEl);
    seg_exp_sum<<<(nEl + 255) / 256, 256, 0, stream>>>(lg, seg, segm, ev, segs, nEl);
  };
  auto run_gru = [&](const float* xin, const float* hprev,
                     const __bf16* Wihb, const __bf16* Whhb,
                     const float* bih, const float* bhh, float* outp, int M) {
    gemm_bias_act<0><<<dim3(6, M / 64), 256, 0, stream>>>(xin, Wihb, bih, gi, 384, HD);
    gemm_bias_act<0><<<dim3(6, M / 64), 256, 0, stream>>>(hprev, Whhb, bhh, gh, 384, HD);
    size_t n = (size_t)M * HD;
    gru_eltwise<<<(unsigned)((n + 255) / 256), 256, 0, stream>>>(gi, gh, hprev, outp, (size_t)M);
  };

  // 1) hx = leaky(x @ lin1_W^T + b)   (+ bf16 mirror for edge gathers)
  lin1_kernel<<<NV, HD, 0, stream>>>(x, lin1_W, lin1_b, hx, hxb);
  // 2) GATEConv
  row_dots<<<NV, HD, 0, stream>>>(hx, g_attr_, nullptr, sdst, nullptr);
  gate_edge_logits<<<NE / 64, 256, 0, stream>>>(hxb, eattr, src, dst, gW1b, g_attl, sdst, elog);
  softmax_seg(elog, dst, NE, NV, eexp);
  fill(agg, 0.f, (size_t)NV * HD);
  gate_msg_scatter<<<NE / 64, 256, 0, stream>>>(hxb, src, dst, gW2b, eexp, segs, agg);
  elu_bias<<<(unsigned)(((size_t)NV * HD + 255) / 256), 256, 0, stream>>>(agg, g_b, hbuf, (size_t)NV * HD);
  // 3) gru0
  run_gru(hbuf, hx, g0Wihb, g0Whhb, g0bih, g0bhh, hx, NV);
  // 4) atom GATConv + GRU x2
  for (int l = 0; l < 2; ++l) {
    gemm_bias_act<0><<<dim3(2, NV / 64), 256, 0, stream>>>(hx, acWb[l], nullptr, hs, HD, HD);
    row_dots<<<NV, HD, 0, stream>>>(hs, acAs[l], acAd[l], d1, d2);
    edge_logits<<<(NE + 255) / 256, 256, 0, stream>>>(d1, d2, src, dst, elog, NE);
    softmax_seg(elog, dst, NE, NV, eexp);
    fill(agg, 0.f, (size_t)NV * HD);
    gat_scatter<<<(unsigned)(((size_t)NE * HD + 255) / 256), 256, 0, stream>>>(hs, src, dst, eexp, segs, agg, NE);
    elu_bias<<<(unsigned)(((size_t)NV * HD + 255) / 256), 256, 0, stream>>>(agg, acB[l], hbuf, (size_t)NV * HD);
    run_gru(hbuf, hx, agWihb[l], agWhhb[l], agBih[l], agBhh[l], hx, NV);
  }
  // 5) molecule readout
  fill(molo, 0.f, (size_t)NG * HD);
  pool_kernel<<<(unsigned)(((size_t)NV * HD + 255) / 256), 256, 0, stream>>>(hx, batch, molo);
  relu_inplace<<<(unsigned)(((size_t)NG * HD + 255) / 256), 256, 0, stream>>>(molo, (size_t)NG * HD);
  gemm_bias_act<0><<<dim3(2, NV / 64), 256, 0, stream>>>(hx, mcWb, nullptr, hs, HD, HD);
  row_dots<<<NV, HD, 0, stream>>>(hs, mcAs, nullptr, d1, nullptr);
  for (int t = 0; t < 3; ++t) {
    gemm_bias_act<0><<<dim3(2, NG / 64), 256, 0, stream>>>(molo, mcWb, nullptr, hd, HD, HD);
    row_dots<<<NG, HD, 0, stream>>>(hd, mcAd, nullptr, d2, nullptr);
    edge_logits<<<(NV + 255) / 256, 256, 0, stream>>>(d1, d2, nullptr, batch, elog, NV);
    softmax_seg(elog, batch, NV, NG, eexp);
    fill(magg, 0.f, (size_t)NG * HD);
    gat_scatter<<<(unsigned)(((size_t)NV * HD + 255) / 256), 256, 0, stream>>>(hs, nullptr, batch, eexp, segs, magg, NV);
    elu_bias<<<(unsigned)(((size_t)NG * HD + 255) / 256), 256, 0, stream>>>(magg, mcB, molh, (size_t)NG * HD);
    run_gru(molh, molo, mgWihb, mgWhhb, mgBih, mgBhh, molo, NG);
  }
  // 6) graph_embed -> comb[:, 0:128]
  gemm_bias_act<0><<<dim3(2, NG / 64), 256, 0, stream>>>(molo, l2Wb, l2b, comb, 384, HD);
  // 7) fingerprint MLP -> comb[:, 128:256]
  gemm_bias_act<1><<<dim3(8, NG / 64), 256, 0, stream>>>(fpr, fpW1b, fpb1, f512, 512, FPDIM);
  bn_inplace<<<(unsigned)(((size_t)NG * 512 + 255) / 256), 256, 0, stream>>>(f512, fpg, fpbe, fpm, fpv, 511, (size_t)NG * 512);
  gemm_bias_act<1><<<dim3(2, NG / 64), 256, 0, stream>>>(f512, fpW2b, fpb2, comb + 128, 384, 512);
  // 8) descriptor MLP -> comb[:, 256:384]
  desc1_kernel<<<NG, 64, 0, stream>>>(descr, deW1, deb1, d64);
  bn_inplace<<<(unsigned)(((size_t)NG * 64 + 255) / 256), 256, 0, stream>>>(d64, deg, debe, dem, dev_, 63, (size_t)NG * 64);
  gemm_bias_act<1><<<dim3(2, NG / 64), 256, 0, stream>>>(d64, deW2b, deb2, comb + 256, 384, 64);
  // 9) fusion + heads
  gemm_bias_act<1><<<dim3(4, NG / 64), 256, 0, stream>>>(comb, fusWb, fusb, shrd, 256, 384);
  heads_kernel<<<NG, 256, 0, stream>>>(shrd, toxW, toxb, regW, regb, (float*)d_out);
}
